// ASMLoc_79413945303624
// MI455X (gfx1250) — matrix operand order repaired and checked by reading, not compiled, not run on hardware
//
#include <hip/hip_runtime.h>
#include <hip/hip_bf16.h>

// ---------------------------------------------------------------------------
// ASM-Loc head for MI455X (gfx1250). All heavy GEMMs use
// v_wmma_f32_16x16x32_f16 (f16 in, f32 accum), wave32, 8 waves / 256-thread
// block on a 64x64 tile. All matrix buffers are padded to tile multiples
// (T: 750->768, K: 50->64) and the workspace is zeroed once per launch, so
// tile staging is branch-free vector loads (global b128 -> LDS b128 for A, b16 for transposed B) and
// fragments are loaded as ds_load_b128 pairs per the CDNA5 VGPR layouts.
// ---------------------------------------------------------------------------

typedef _Float16 half_t;
typedef __attribute__((ext_vector_type(16))) _Float16 v16h;
typedef __attribute__((ext_vector_type(8)))  _Float16 v8h;
typedef __attribute__((ext_vector_type(8)))  float    v8f;

#define Bn     16
#define Tn     750
#define TP     768      // Tn padded to multiple of 64
#define Cn     2048
#define Kn     50
#define KP     64       // Kn padded to multiple of 64
#define Hn     8
#define DHn    256
#define NCLSn  21
#define FG_TOPKn 93
#define BG_TOPKn 250
#define SCALEf 0.0625f  // DH^-0.5 = 1/16

// vectorized epilogue stores (8 consecutive elements)
__device__ inline void store8(float* p, const float* v) {
    v8f t;
#pragma unroll
    for (int r = 0; r < 8; ++r) t[r] = v[r];
    *(v8f*)p = t;
}
__device__ inline void store8(half_t* p, const float* v) {
    v8h t;
#pragma unroll
    for (int r = 0; r < 8; ++r) t[r] = (half_t)v[r];
    *(v8h*)p = t;
}

// ---------------------------------------------------------------------------
// Tiled WMMA GEMM:  C[M,N] = act( alpha * A[M,K] * B[K,N] + bias[M] )
// A,B row-major f16, dims padded (grid covers padded M/N; Kd multiple of 32;
// padding regions of A/B are zero). Stores guarded by real M,N.
// z decomposed as (z/zdiv, z%zdiv) for (batch, head) strides.
// ---------------------------------------------------------------------------
template<typename OutT, bool RELU, bool TRANS_STORE>
__global__ void __launch_bounds__(256)
gemm_wmma_kernel(const half_t* __restrict__ Ag, long long lda, long long sA1, long long sA2,
                 const half_t* __restrict__ Bg, long long ldb, long long sB1, long long sB2,
                 OutT* __restrict__ Cg, long long ldc, long long sC1, long long sC2,
                 int M, int N, int Kd, int zdiv,
                 const float* __restrict__ bias, float alpha)
{
    __shared__ half_t As [64][40];   // rows x K (row stride 80B, 16B aligned)
    __shared__ half_t BsT[64][40];   // cols x K (transposed stage)

    const int tid  = threadIdx.x;
    const int lane = tid & 31;
    const int wid  = tid >> 5;      // 0..7
    const int wm   = wid >> 1;      // 0..3 -> 16-row strip
    const int wn   = wid & 1;       // 0..1 -> 32-col strip

    const int z  = blockIdx.z;
    const int b1 = z / zdiv;
    const int b2 = z % zdiv;
    Ag += (long long)b1 * sA1 + (long long)b2 * sA2;
    Bg += (long long)b1 * sB1 + (long long)b2 * sB2;
    Cg += (long long)b1 * sC1 + (long long)b2 * sC2;

    const int rowBlk = blockIdx.y * 64;
    const int colBlk = blockIdx.x * 64;

    v8f acc0, acc1;
#pragma unroll
    for (int r = 0; r < 8; ++r) { acc0[r] = 0.f; acc1[r] = 0.f; }

    const int arow = tid >> 2;          // 0..63
    const int acol = (tid & 3) * 8;     // 0,8,16,24
    const int brow = tid >> 3;          // 0..31  (k)
    const int bcol = (tid & 7) * 8;     // 0..56  (col)

    // fragment addressing (CDNA5 wave32 layouts)
    const int frow  = wm * 16 + (lane & 15);      // A row in tile
    const int kselA = (lane >= 16) ? 1 : 0;       // A: +8 halves
    const int kselB = (lane >= 16) ? 1 : 0;       // B: +8 halves
    const int c0    = wn * 32 + (lane & 15);      // B col (first 16x16)

    for (int kt = 0; kt < Kd; kt += 32) {
        __syncthreads();
        // stage A: one global b128 load + one ds b128 store per thread
        {
            const v8h g = *(const v8h*)&Ag[(long long)(rowBlk + arow) * lda + kt + acol];
            *(v8h*)&As[arow][acol] = g;
        }
        // stage B transposed: one global b128 load + 8 b16 stores per thread
        {
            const v8h g = *(const v8h*)&Bg[(long long)(kt + brow) * ldb + colBlk + bcol];
#pragma unroll
            for (int e = 0; e < 8; ++e) BsT[bcol + e][brow] = g[e];
        }
        __syncthreads();

        if (kt + 32 < Kd) {  // gfx1250 global_prefetch_b8 on next K tile
            __builtin_prefetch(&Ag[(long long)(rowBlk + arow) * lda + kt + 32 + acol], 0, 3);
            __builtin_prefetch(&Bg[(long long)(kt + 32 + brow) * ldb + colBlk + bcol], 0, 3);
        }

        // A fragment: lane holds K = {kadd..kadd+7, kadd+16..kadd+23}
        const v8h* ap = (const v8h*)&As[frow][0];
        const v16h afr = __builtin_shufflevector(ap[kselA], ap[kselA + 2],
                                                 0,1,2,3,4,5,6,7,8,9,10,11,12,13,14,15);
        // B fragments: lane holds col c0(+16), K = {kadd..kadd+7, kadd+16..kadd+23}
        const v8h* bp0 = (const v8h*)&BsT[c0][0];
        const v8h* bp1 = (const v8h*)&BsT[c0 + 16][0];
        const v16h bf0 = __builtin_shufflevector(bp0[kselB], bp0[kselB + 2],
                                                 0,1,2,3,4,5,6,7,8,9,10,11,12,13,14,15);
        const v16h bf1 = __builtin_shufflevector(bp1[kselB], bp1[kselB + 2],
                                                 0,1,2,3,4,5,6,7,8,9,10,11,12,13,14,15);

        acc0 = __builtin_amdgcn_wmma_f32_16x16x32_f16(false, afr, false, bf0, (short)0, acc0, false, false);
        acc1 = __builtin_amdgcn_wmma_f32_16x16x32_f16(false, afr, false, bf1, (short)0, acc1, false, false);
    }

    // epilogue: C/D layout — VGPR r: M = r (+8 for lanes>=16); N = lane&15
    const int m0 = rowBlk + wm * 16 + ((lane >= 16) ? 8 : 0);
#pragma unroll
    for (int sub = 0; sub < 2; ++sub) {
        const v8f acc = sub ? acc1 : acc0;
        const int n = colBlk + wn * 32 + sub * 16 + (lane & 15);
        if (n >= N) continue;
        float vals[8];
#pragma unroll
        for (int r = 0; r < 8; ++r) {
            float v = acc[r] * alpha + (bias ? bias[m0 + r] : 0.f);
            if (RELU) v = fmaxf(v, 0.f);
            vals[r] = v;
        }
        if (TRANS_STORE) {
            // 8 consecutive m -> one vector store when fully in range
            if (m0 + 8 <= M) store8(&Cg[(long long)n * ldc + m0], vals);
            else {
#pragma unroll
                for (int r = 0; r < 8; ++r)
                    if (m0 + r < M) Cg[(long long)n * ldc + m0 + r] = (OutT)vals[r];
            }
        } else {
#pragma unroll
            for (int r = 0; r < 8; ++r)
                if (m0 + r < M) Cg[(long long)(m0 + r) * ldc + n] = (OutT)vals[r];
        }
    }
}

// ---------------------------------------------------------------------------
// Support kernels
// ---------------------------------------------------------------------------
__global__ void cvt_f32_to_f16_kernel(const float* __restrict__ in, half_t* __restrict__ out, long long n)
{
    long long i = (long long)blockIdx.x * 256 + threadIdx.x;
    if (i < n) out[i] = (half_t)in[i];
}

// emb_w [C,C,3] -> [C, 3C] with col = d*C + c
__global__ void build_embw_kernel(const float* __restrict__ w, half_t* __restrict__ out)
{
    long long i = (long long)blockIdx.x * 256 + threadIdx.x;
    const long long total = (long long)Cn * 3 * Cn;
    if (i >= total) return;
    int o = (int)(i / (3 * Cn));
    int r = (int)(i % (3 * Cn));
    int d = r / Cn, c = r % Cn;
    out[i] = (half_t)w[((long long)o * Cn + c) * 3 + d];
}

// im2col k=3 pad=1 into padded [B, 3C, TP]: xcol[b, d*C+c, t] = x[b, t+d-1, c]
__global__ void build_xcol_kernel(const float* __restrict__ x, half_t* __restrict__ out)
{
    long long i = (long long)blockIdx.x * 256 + threadIdx.x;
    const long long total = (long long)Bn * 3 * Cn * TP;
    if (i >= total) return;
    int t = (int)(i % TP);
    int r = (int)((i / TP) % (3LL * Cn));
    int b = (int)(i / ((long long)TP * 3 * Cn));
    int d = r / Cn, c = r % Cn;
    int tt = t + d - 1;
    float v = (t < Tn && tt >= 0 && tt < Tn) ? x[((long long)b * Tn + tt) * Cn + c] : 0.f;
    out[i] = (half_t)v;
}

// proposal coverage bitmask: bit k set iff start<=t<=end and k<count
__global__ void build_pbits_kernel(const int* __restrict__ bbox, const int* __restrict__ cnt,
                                   unsigned long long* __restrict__ pbits)
{
    long long i = (long long)blockIdx.x * 256 + threadIdx.x;
    if (i >= (long long)Bn * Tn) return;
    int t = (int)(i % Tn), b = (int)(i / Tn);
    int c = cnt[b]; if (c > Kn) c = Kn;
    unsigned long long bits = 0ull;
    for (int k = 0; k < c; ++k) {
        int s = bbox[((long long)b * Kn + k) * 2 + 0];
        int e = bbox[((long long)b * Kn + k) * 2 + 1];
        if (t >= s && t <= e) bits |= (1ull << k);
    }
    pbits[i] = bits;
}

// masked row softmax over sim[b,h,n,:], nan_to_num, write attn^T (f16)
__global__ void __launch_bounds__(256)
softmax_intra_kernel(const float* __restrict__ sim, const unsigned long long* __restrict__ pbits,
                     half_t* __restrict__ attnT)
{
    const int n = blockIdx.x, h = blockIdx.y, b = blockIdx.z, tid = threadIdx.x;
    __shared__ float red[256];
    const float* row = sim + (((long long)(b * Hn + h)) * TP + n) * TP;
    half_t* outc = attnT + ((long long)(b * Hn + h)) * TP * TP + n;
    const unsigned long long pn = pbits[(long long)b * Tn + n];

    float mx = -3.0e38f;
    for (int m = tid; m < Tn; m += 256)
        if (pn & pbits[(long long)b * Tn + m]) mx = fmaxf(mx, row[m]);
    red[tid] = mx; __syncthreads();
    for (int s = 128; s > 0; s >>= 1) { if (tid < s) red[tid] = fmaxf(red[tid], red[tid + s]); __syncthreads(); }
    mx = red[0]; __syncthreads();

    float sum = 0.f;
    for (int m = tid; m < Tn; m += 256)
        if (pn & pbits[(long long)b * Tn + m]) sum += __expf(row[m] - mx);
    red[tid] = sum; __syncthreads();
    for (int s = 128; s > 0; s >>= 1) { if (tid < s) red[tid] += red[tid + s]; __syncthreads(); }
    sum = red[0];

    const bool dead = (mx < -1.0e38f) || !(sum > 0.f);   // fully-masked row -> zeros
    const float inv = dead ? 0.f : 1.f / sum;
    for (int m = tid; m < Tn; m += 256) {
        float v = 0.f;
        if (!dead && (pn & pbits[(long long)b * Tn + m])) v = __expf(row[m] - mx) * inv;
        outc[(long long)m * TP] = (half_t)v;
    }
}

// inter-segment masked softmax (mask = valid outer product), write attn^T f16
__global__ void __launch_bounds__(64)
softmax_inter_kernel(const float* __restrict__ sim, const int* __restrict__ pcount,
                     half_t* __restrict__ attnT)
{
    const int i = blockIdx.x, h = blockIdx.y, b = blockIdx.z, tid = threadIdx.x;
    __shared__ float red[64];
    int cnt = pcount[b]; if (cnt > Kn) cnt = Kn;
    const float* row = sim + (((long long)(b * Hn + h)) * KP + i) * KP;
    half_t* outc = attnT + ((long long)(b * Hn + h)) * KP * KP + i;
    const bool rowValid = (i < cnt);

    float mx = -3.0e38f;
    if (rowValid) for (int j = tid; j < cnt; j += 64) mx = fmaxf(mx, row[j]);
    red[tid] = mx; __syncthreads();
    for (int s = 32; s > 0; s >>= 1) { if (tid < s) red[tid] = fmaxf(red[tid], red[tid + s]); __syncthreads(); }
    mx = red[0]; __syncthreads();

    float sum = 0.f;
    if (rowValid) for (int j = tid; j < cnt; j += 64) sum += __expf(row[j] - mx);
    red[tid] = sum; __syncthreads();
    for (int s = 32; s > 0; s >>= 1) { if (tid < s) red[tid] += red[tid + s]; __syncthreads(); }
    sum = red[0];

    const bool dead = !rowValid || !(sum > 0.f);
    const float inv = dead ? 0.f : 1.f / sum;
    for (int j = tid; j < Kn; j += 64) {
        float v = (!dead && j < cnt) ? __expf(row[j] - mx) * inv : 0.f;
        outc[(long long)j * KP] = (half_t)v;
    }
}

// out = embIn + BN(o)   (buffers padded [B,C,TP]; only t<Tn touched)
__global__ void bn_res_kernel(const float* __restrict__ embIn, const float* __restrict__ o,
                              const float* __restrict__ g, const float* __restrict__ bb,
                              const float* __restrict__ mm, const float* __restrict__ vv,
                              float* __restrict__ embOut)
{
    long long i = (long long)blockIdx.x * 256 + threadIdx.x;
    const long long total = (long long)Bn * Cn * TP;
    if (i >= total) return;
    int t = (int)(i % TP);
    if (t >= Tn) return;
    int c = (int)((i / TP) % Cn);
    embOut[i] = embIn[i] + g[c] * (o[i] - mm[c]) * rsqrtf(vv[c] + 1e-5f) + bb[c];
}

// segment pooling: seg[b,c,k] = mean over covered t (0 if invalid k / padding)
__global__ void pool_seg_kernel(const float* __restrict__ emb2, const int* __restrict__ bbox,
                                const int* __restrict__ pcount, half_t* __restrict__ seg)
{
    long long i = (long long)blockIdx.x * 256 + threadIdx.x;
    const long long total = (long long)Bn * Cn * KP;
    if (i >= total) return;
    int k = (int)(i % KP);
    int c = (int)((i / KP) % Cn);
    int b = (int)(i / ((long long)KP * Cn));
    float val = 0.f;
    if (k < Kn && k < pcount[b]) {
        int s = bbox[((long long)b * Kn + k) * 2 + 0];
        int e = bbox[((long long)b * Kn + k) * 2 + 1];
        const float* src = emb2 + ((long long)b * Cn + c) * TP;
        float acc = 0.f;
        for (int t = s; t <= e; ++t) acc += src[t];
        val = acc / (float)(e - s + 1);
    }
    seg[i] = (half_t)val;
}

// scatter back: sc[b,c,t] = (sum over covering k of ctxe[b,c,k]) / cover(t)
__global__ void scatter_seg_kernel(const half_t* __restrict__ ctxe,
                                   const unsigned long long* __restrict__ pbits,
                                   half_t* __restrict__ sc)
{
    long long i = (long long)blockIdx.x * 256 + threadIdx.x;
    const long long total = (long long)Bn * Cn * Tn;
    if (i >= total) return;
    int t = (int)(i % Tn);
    int c = (int)((i / Tn) % Cn);
    int b = (int)(i / ((long long)Tn * Cn));
    unsigned long long pb = pbits[(long long)b * Tn + t];
    int cov = __popcll(pb);
    float sum = 0.f;
    while (pb) {
        int k = __ffsll((long long)pb) - 1;
        pb &= pb - 1;
        sum += (float)ctxe[((long long)b * Cn + c) * KP + k];
    }
    sc[((long long)b * Cn + c) * TP + t] = (half_t)(cov ? sum / (float)cov : 0.f);
}

// per-(b,t) heads: att softmax, cas softmax, fg/bg cas (+raw for top-k), unc
__global__ void __launch_bounds__(128)
heads_kernel(const float* __restrict__ emb3,
             const float* __restrict__ att_w, const float* __restrict__ att_b,
             const float* __restrict__ cls_w, const float* __restrict__ cls_b,
             const float* __restrict__ unc_w,
             float* __restrict__ out_att, float* __restrict__ out_cas,
             float* __restrict__ out_fgsm, float* __restrict__ out_bgsm,
             float* __restrict__ out_unc,
             float* __restrict__ fg_raw, float* __restrict__ bg_raw)
{
    const int t = blockIdx.x, b = blockIdx.y, tid = threadIdx.x;
    __shared__ float red[128];
    __shared__ float vals[24];   // 0,1=att  2=unc  3..23=cas
    const float* col = emb3 + (long long)b * Cn * TP + t;   // channel stride = TP

    for (int o = 0; o < 24; ++o) {
        float s = 0.f;
        for (int c = tid; c < Cn; c += 128) {
            float x = col[(long long)c * TP];
            float w;
            if (o < 2)       w = att_w[(long long)o * Cn + c];
            else if (o == 2) w = unc_w[c];
            else             w = cls_w[(long long)c * NCLSn + (o - 3)];
            s += x * w;
        }
        red[tid] = s; __syncthreads();
        for (int st = 64; st > 0; st >>= 1) { if (tid < st) red[tid] += red[tid + st]; __syncthreads(); }
        if (tid == 0) vals[o] = red[0] + (o < 2 ? att_b[o] : (o == 2 ? 0.f : cls_b[o - 3]));
        __syncthreads();
    }

    if (tid == 0) {
        const long long bt = (long long)b * Tn + t;
        float a0 = vals[0], a1 = vals[1];
        float am = fmaxf(a0, a1);
        float e0 = __expf(a0 - am), e1 = __expf(a1 - am);
        float ainv = 1.f / (e0 + e1);
        float p0 = e0 * ainv, p1 = e1 * ainv;
        out_att[bt * 2 + 0] = p0;
        out_att[bt * 2 + 1] = p1;
        out_unc[bt] = vals[2];
        float cas[NCLSn], fgv[NCLSn], bgv[NCLSn];
        float mc = -3.0e38f;
        for (int j = 0; j < NCLSn; ++j) { cas[j] = vals[3 + j]; mc = fmaxf(mc, cas[j]); }
        float cs = 0.f;
        for (int j = 0; j < NCLSn; ++j) cs += __expf(cas[j] - mc);
        float cinv = 1.f / cs;
        for (int j = 0; j < NCLSn; ++j) out_cas[bt * NCLSn + j] = __expf(cas[j] - mc) * cinv;
        float mf = -3.0e38f, mb = -3.0e38f;
        for (int j = 0; j < NCLSn; ++j) {
            fgv[j] = cas[j] * p0; bgv[j] = cas[j] * p1;
            fg_raw[((long long)b * NCLSn + j) * Tn + t] = fgv[j];
            bg_raw[((long long)b * NCLSn + j) * Tn + t] = bgv[j];
            mf = fmaxf(mf, fgv[j]); mb = fmaxf(mb, bgv[j]);
        }
        float fs = 0.f, bs = 0.f;
        for (int j = 0; j < NCLSn; ++j) { fs += __expf(fgv[j] - mf); bs += __expf(bgv[j] - mb); }
        float fi = 1.f / fs, bi = 1.f / bs;
        for (int j = 0; j < NCLSn; ++j) {
            out_fgsm[bt * NCLSn + j] = __expf(fgv[j] - mf) * fi;
            out_bgsm[bt * NCLSn + j] = __expf(bgv[j] - mb) * bi;
        }
    }
}

// bitonic sort (ascending, 1024 padded with -inf) -> mean of top 'topk'
__global__ void __launch_bounds__(256)
topk_mean_kernel(const float* __restrict__ raw, float* __restrict__ outv, int topk)
{
    const int cls = blockIdx.x, b = blockIdx.y, tid = threadIdx.x;
    __shared__ float s[1024];
    __shared__ float red[256];
    const float* src = raw + ((long long)b * NCLSn + cls) * Tn;
    for (int i = tid; i < 1024; i += 256) s[i] = (i < Tn) ? src[i] : -3.0e38f;
    __syncthreads();
    for (int k = 2; k <= 1024; k <<= 1) {
        for (int j = k >> 1; j > 0; j >>= 1) {
            for (int i = tid; i < 1024; i += 256) {
                int ixj = i ^ j;
                if (ixj > i) {
                    bool up = ((i & k) == 0);
                    float a = s[i], c = s[ixj];
                    if (up ? (a > c) : (a < c)) { s[i] = c; s[ixj] = a; }
                }
            }
            __syncthreads();
        }
    }
    float sum = 0.f;
    for (int i = 1024 - topk + tid; i < 1024; i += 256) sum += s[i];
    red[tid] = sum; __syncthreads();
    for (int st = 128; st > 0; st >>= 1) { if (tid < st) red[tid] += red[tid + st]; __syncthreads(); }
    if (tid == 0) outv[(long long)b * NCLSn + cls] = red[0] / (float)topk;
}

__global__ void cls_softmax_kernel(const float* __restrict__ in, float* __restrict__ out)
{
    if (threadIdx.x != 0) return;
    const int b = blockIdx.x;
    float m = -3.0e38f;
    for (int j = 0; j < NCLSn; ++j) m = fmaxf(m, in[(long long)b * NCLSn + j]);
    float s = 0.f;
    for (int j = 0; j < NCLSn; ++j) s += __expf(in[(long long)b * NCLSn + j] - m);
    float inv = 1.f / s;
    for (int j = 0; j < NCLSn; ++j) out[(long long)b * NCLSn + j] = __expf(in[(long long)b * NCLSn + j] - m) * inv;
}

// ---------------------------------------------------------------------------
// Host orchestration
// ---------------------------------------------------------------------------
extern "C" void kernel_launch(void* const* d_in, const int* in_sizes, int n_in,
                              void* d_out, int out_size, void* d_ws, size_t ws_size,
                              hipStream_t stream)
{
    (void)in_sizes; (void)n_in; (void)out_size; (void)ws_size;

    const float* in_feat = (const float*)d_in[0];
    const int*   bbox    = (const int*)d_in[1];
    const int*   pcount  = (const int*)d_in[2];
    const float* emb_w   = (const float*)d_in[3];
    const float* emb_b   = (const float*)d_in[4];
    const float* i_qw = (const float*)d_in[5];  const float* i_qb = (const float*)d_in[6];
    const float* i_kw = (const float*)d_in[7];  const float* i_kb = (const float*)d_in[8];
    const float* i_vw = (const float*)d_in[9];  const float* i_vb = (const float*)d_in[10];
    const float* i_ow = (const float*)d_in[11]; const float* i_ob = (const float*)d_in[12];
    const float* i_bng = (const float*)d_in[13]; const float* i_bnb = (const float*)d_in[14];
    const float* i_bnm = (const float*)d_in[15]; const float* i_bnv = (const float*)d_in[16];
    const float* e_qw = (const float*)d_in[17]; const float* e_qb = (const float*)d_in[18];
    const float* e_kw = (const float*)d_in[19]; const float* e_kb = (const float*)d_in[20];
    const float* e_vw = (const float*)d_in[21]; const float* e_vb = (const float*)d_in[22];
    const float* e_ow = (const float*)d_in[23]; const float* e_ob = (const float*)d_in[24];
    const float* e_bng = (const float*)d_in[25]; const float* e_bnb = (const float*)d_in[26];
    const float* e_bnm = (const float*)d_in[27]; const float* e_bnv = (const float*)d_in[28];
    const float* att_w = (const float*)d_in[29]; const float* att_b = (const float*)d_in[30];
    const float* cls_w = (const float*)d_in[31]; const float* cls_b = (const float*)d_in[32];
    const float* unc_w = (const float*)d_in[33];

    // output layout (concatenated, f32)
    float* out        = (float*)d_out;
    float* out_fg_cls = out;
    float* out_bg_cls = out + 336;
    float* out_att    = out + 672;
    float* out_cas    = out_att + (long long)Bn*Tn*2;
    float* out_fgsm   = out_cas + (long long)Bn*Tn*NCLSn;
    float* out_bgsm   = out_fgsm + (long long)Bn*Tn*NCLSn;
    float* out_unc    = out_bgsm + (long long)Bn*Tn*NCLSn;

    // workspace bump allocator (padded layouts)
    char* wsb = (char*)d_ws;
    size_t off = 0;
    auto alloc = [&](size_t bytes) -> char* {
        char* p = wsb + off;
        off += (bytes + 255) & ~(size_t)255;
        return p;
    };
    const long long CTP  = (long long)Cn * TP;
    const long long BCTP = (long long)Bn * CTP;
    const long long CC   = (long long)Cn * Cn;
    const long long CKP  = (long long)Cn * KP;

    half_t* wq_i  = (half_t*)alloc(CC * 2);
    half_t* wk_i  = (half_t*)alloc(CC * 2);
    half_t* wv_i  = (half_t*)alloc(CC * 2);
    half_t* wo_i  = (half_t*)alloc(CC * 2);
    half_t* wq_e  = (half_t*)alloc(CC * 2);
    half_t* wk_e  = (half_t*)alloc(CC * 2);
    half_t* wv_e  = (half_t*)alloc(CC * 2);
    half_t* wo_e  = (half_t*)alloc(CC * 2);
    half_t* embW16 = (half_t*)alloc((long long)Cn * 3 * Cn * 2);
    half_t* xcol16 = (half_t*)alloc((long long)Bn * 3 * Cn * TP * 2);
    half_t* emb16  = (half_t*)alloc(BCTP * 2);
    half_t* qT16   = (half_t*)alloc((long long)Bn * TP * Cn * 2);
    half_t* k16    = (half_t*)alloc(BCTP * 2);
    half_t* v16    = (half_t*)alloc(BCTP * 2);
    half_t* attnT16 = (half_t*)alloc((long long)Bn * Hn * TP * TP * 2);
    half_t* ctx16  = (half_t*)alloc(BCTP * 2);
    half_t* seg16  = (half_t*)alloc((long long)Bn * CKP * 2);
    half_t* qTe16  = (half_t*)alloc((long long)Bn * KP * Cn * 2);
    half_t* ke16   = (half_t*)alloc((long long)Bn * CKP * 2);
    half_t* ve16   = (half_t*)alloc((long long)Bn * CKP * 2);
    half_t* attnTe16 = (half_t*)alloc((long long)Bn * Hn * KP * KP * 2);
    half_t* ctxe16 = (half_t*)alloc((long long)Bn * CKP * 2);
    half_t* sc16   = (half_t*)alloc(BCTP * 2);

    float* emb_f  = (float*)alloc(BCTP * 4);
    float* o_f    = (float*)alloc(BCTP * 4);
    float* emb2_f = (float*)alloc(BCTP * 4);
    float* emb3_f = (float*)alloc(BCTP * 4);
    float* sim_f  = (float*)alloc((long long)Bn * Hn * TP * TP * 4);
    float* sime_f = (float*)alloc((long long)Bn * Hn * KP * KP * 4);
    float* fg_raw = (float*)alloc((long long)Bn * NCLSn * Tn * 4);
    float* bg_raw = (float*)alloc((long long)Bn * NCLSn * Tn * 4);
    float* tmp_fg = (float*)alloc((long long)Bn * NCLSn * 4);
    float* tmp_bg = (float*)alloc((long long)Bn * NCLSn * 4);
    unsigned long long* pbits = (unsigned long long*)alloc((long long)Bn * Tn * 8);

    // zero the workspace so every padding region reads as 0
    hipMemsetAsync(d_ws, 0, off, stream);

    auto cvt = [&](const float* src, half_t* dst, long long n) {
        unsigned blocks = (unsigned)((n + 255) / 256);
        cvt_f32_to_f16_kernel<<<blocks, 256, 0, stream>>>(src, dst, n);
    };

    // ---- weight / input prep -------------------------------------------------
    build_embw_kernel<<<(unsigned)(((long long)Cn*3*Cn + 255)/256), 256, 0, stream>>>(emb_w, embW16);
    build_xcol_kernel<<<(unsigned)(((long long)Bn*3*Cn*TP + 255)/256), 256, 0, stream>>>(in_feat, xcol16);
    cvt(i_qw, wq_i, CC); cvt(i_kw, wk_i, CC); cvt(i_vw, wv_i, CC); cvt(i_ow, wo_i, CC);
    cvt(e_qw, wq_e, CC); cvt(e_kw, wk_e, CC); cvt(e_vw, wv_e, CC); cvt(e_ow, wo_e, CC);
    build_pbits_kernel<<<(unsigned)(((long long)Bn*Tn + 255)/256), 256, 0, stream>>>(bbox, pcount, pbits);

    const dim3 blk(256);
    const dim3 gProj(TP/64, Cn/64, Bn);       // 12 x 32 x 16
    // ---- emb = relu(conv1d(x) + b)  as [C,3C] x [3C,TP] ----------------------
    gemm_wmma_kernel<float, true, false><<<gProj, blk, 0, stream>>>(
        embW16, 3*Cn, 0, 0,  xcol16, TP, (long long)3*Cn*TP, 0,
        emb_f, TP, CTP, 0,  Cn, Tn, 3*Cn, 1, emb_b, 1.0f);
    cvt(emb_f, emb16, BCTP);

    // ---- intra q/k/v projections --------------------------------------------
    gemm_wmma_kernel<half_t, false, true><<<gProj, blk, 0, stream>>>(   // q^T [B,TP,C]
        wq_i, Cn, 0, 0,  emb16, TP, CTP, 0,
        qT16, Cn, (long long)TP*Cn, 0,  Cn, Tn, Cn, 1, i_qb, 1.0f);
    gemm_wmma_kernel<half_t, false, false><<<gProj, blk, 0, stream>>>(  // k [B,C,TP]
        wk_i, Cn, 0, 0,  emb16, TP, CTP, 0,
        k16, TP, CTP, 0,  Cn, Tn, Cn, 1, i_kb, 1.0f);
    gemm_wmma_kernel<half_t, false, false><<<gProj, blk, 0, stream>>>(  // v [B,C,TP]
        wv_i, Cn, 0, 0,  emb16, TP, CTP, 0,
        v16, TP, CTP, 0,  Cn, Tn, Cn, 1, i_vb, 1.0f);

    // ---- scores: sim[b,h] = SCALE * q^T(TxDH) x k(DHxT) ----------------------
    const dim3 gSim(TP/64, TP/64, Bn * Hn);
    gemm_wmma_kernel<float, false, false><<<gSim, blk, 0, stream>>>(
        qT16, Cn, (long long)TP*Cn, DHn,
        k16, TP, CTP, (long long)DHn*TP,
        sim_f, TP, (long long)Hn*TP*TP, (long long)TP*TP,
        Tn, Tn, DHn, Hn, nullptr, SCALEf);
    softmax_intra_kernel<<<dim3(Tn, Hn, Bn), 256, 0, stream>>>(sim_f, pbits, attnT16);

    // ---- context: ctx[b,h] = v(DHxT) x attn^T(TxT) ---------------------------
    const dim3 gCtx(TP/64, DHn/64, Bn * Hn);
    gemm_wmma_kernel<half_t, false, false><<<gCtx, blk, 0, stream>>>(
        v16, TP, CTP, (long long)DHn*TP,
        attnT16, TP, (long long)Hn*TP*TP, (long long)TP*TP,
        ctx16, TP, CTP, (long long)DHn*TP,
        DHn, Tn, TP, Hn, nullptr, 1.0f);

    // ---- intra out-proj + BN residual ---------------------------------------
    gemm_wmma_kernel<float, false, false><<<gProj, blk, 0, stream>>>(
        wo_i, Cn, 0, 0,  ctx16, TP, CTP, 0,
        o_f, TP, CTP, 0,  Cn, Tn, Cn, 1, i_ob, 1.0f);
    bn_res_kernel<<<(unsigned)((BCTP + 255)/256), 256, 0, stream>>>(
        emb_f, o_f, i_bng, i_bnb, i_bnm, i_bnv, emb2_f);

    // ---- segment pooling -----------------------------------------------------
    pool_seg_kernel<<<(unsigned)(((long long)Bn*CKP + 255)/256), 256, 0, stream>>>(
        emb2_f, bbox, pcount, seg16);

    // ---- inter attention on segments [C,KP] ----------------------------------
    const dim3 gProjE(KP/64, Cn/64, Bn);      // 1 x 32 x 16
    gemm_wmma_kernel<half_t, false, true><<<gProjE, blk, 0, stream>>>(   // qTe [B,KP,C]
        wq_e, Cn, 0, 0,  seg16, KP, CKP, 0,
        qTe16, Cn, (long long)KP*Cn, 0,  Cn, Kn, Cn, 1, e_qb, 1.0f);
    gemm_wmma_kernel<half_t, false, false><<<gProjE, blk, 0, stream>>>(
        wk_e, Cn, 0, 0,  seg16, KP, CKP, 0,
        ke16, KP, CKP, 0,  Cn, Kn, Cn, 1, e_kb, 1.0f);
    gemm_wmma_kernel<half_t, false, false><<<gProjE, blk, 0, stream>>>(
        wv_e, Cn, 0, 0,  seg16, KP, CKP, 0,
        ve16, KP, CKP, 0,  Cn, Kn, Cn, 1, e_vb, 1.0f);

    const dim3 gSimE(1, 1, Bn * Hn);
    gemm_wmma_kernel<float, false, false><<<gSimE, blk, 0, stream>>>(
        qTe16, Cn, (long long)KP*Cn, DHn,
        ke16, KP, CKP, (long long)DHn*KP,
        sime_f, KP, (long long)Hn*KP*KP, (long long)KP*KP,
        Kn, Kn, DHn, Hn, nullptr, SCALEf);
    softmax_inter_kernel<<<dim3(Kn, Hn, Bn), 64, 0, stream>>>(sime_f, pcount, attnTe16);

    const dim3 gCtxE(1, DHn/64, Bn * Hn);
    gemm_wmma_kernel<half_t, false, false><<<gCtxE, blk, 0, stream>>>(
        ve16, KP, CKP, (long long)DHn*KP,
        attnTe16, KP, (long long)Hn*KP*KP, (long long)KP*KP,
        ctxe16, KP, CKP, (long long)DHn*KP,
        DHn, Kn, KP, Hn, nullptr, 1.0f);

    // ---- scatter back, out-proj, BN residual --------------------------------
    scatter_seg_kernel<<<(unsigned)(((long long)Bn*Cn*Tn + 255)/256), 256, 0, stream>>>(
        ctxe16, pbits, sc16);
    gemm_wmma_kernel<float, false, false><<<gProj, blk, 0, stream>>>(
        wo_e, Cn, 0, 0,  sc16, TP, CTP, 0,
        o_f, TP, CTP, 0,  Cn, Tn, Cn, 1, e_ob, 1.0f);
    bn_res_kernel<<<(unsigned)((BCTP + 255)/256), 256, 0, stream>>>(
        emb2_f, o_f, e_bng, e_bnb, e_bnm, e_bnv, emb3_f);

    // ---- heads, top-k, class softmaxes --------------------------------------
    heads_kernel<<<dim3(Tn, Bn), 128, 0, stream>>>(
        emb3_f, att_w, att_b, cls_w, cls_b, unc_w,
        out_att, out_cas, out_fgsm, out_bgsm, out_unc, fg_raw, bg_raw);
    topk_mean_kernel<<<dim3(NCLSn, Bn), 256, 0, stream>>>(fg_raw, tmp_fg, FG_TOPKn);
    topk_mean_kernel<<<dim3(NCLSn, Bn), 256, 0, stream>>>(bg_raw, tmp_bg, BG_TOPKn);
    cls_softmax_kernel<<<Bn, 32, 0, stream>>>(tmp_fg, out_fg_cls);
    cls_softmax_kernel<<<Bn, 32, 0, stream>>>(tmp_bg, out_bg_cls);
}
